// MultiHeadAttention_36807869727470
// MI455X (gfx1250) — matrix operand order repaired and checked by reading, not compiled, not run on hardware
//
#include <hip/hip_runtime.h>
#include <hip/hip_bf16.h>

// MI455X (gfx1250) multi-head causal attention forward.
//  B=256, T=256, C=384, H=6, D=64.
//  All matmuls via v_wmma_f32_16x16x32_f16 (f16 in, f32 accumulate).
//  Attention K/V tiles staged into LDS with the Tensor Data Mover
//  (tensor_load_to_lds + s_wait_tensorcnt), with TDM hardware LDS padding
//  chosen to make the ds_load_b128 B-fragment reads bank-conflict free.
//  d_ws layout (bytes): [q f16 48MB][k f16 48MB][vT f16 48MB][attn-out f16 48MB]

#define H_ 6
#define D_ 64
#define C_ 384
#define T_ 256
#define B_ 256
#define BT_ (B_ * T_)        // 65536 rows
#define PSTR 264             // padded LDS row stride (halfs) for P transpose
#define RSTK 72              // K tile LDS row stride (halfs): 64 + 8 pad
#define RSTV 264             // V^T tile LDS row stride (halfs): 256 + 8 pad

typedef __attribute__((ext_vector_type(16))) _Float16 v16h;
typedef __attribute__((ext_vector_type(8)))  _Float16 v8h;
typedef __attribute__((ext_vector_type(8)))  float    v8f;
typedef __attribute__((ext_vector_type(4)))  unsigned int v4u;
typedef __attribute__((ext_vector_type(4)))  int      v4i;
typedef __attribute__((ext_vector_type(8)))  int      v8i;

#ifndef __has_builtin
#define __has_builtin(x) 0
#endif
#if __has_builtin(__builtin_amdgcn_tensor_load_to_lds) && \
    __has_builtin(__builtin_amdgcn_s_wait_tensorcnt)
#define USE_TDM 1
#else
#define USE_TDM 0
#endif

__device__ __forceinline__ v8f wmma_f16(v16h a, v16h b, v8f c) {
  // 8 args: (neg_a, A, neg_b, B, c_mod, C, reuse_a, reuse_b)
  return __builtin_amdgcn_wmma_f32_16x16x32_f16(false, a, false, b, (short)0, c,
                                                false, false);
}

// Pack 16 f32 into a v16h in order.
__device__ __forceinline__ v16h pack16(float4 f0, float4 f1, float4 f2, float4 f3) {
  v16h a;
  a[0]=(_Float16)f0.x;  a[1]=(_Float16)f0.y;  a[2]=(_Float16)f0.z;  a[3]=(_Float16)f0.w;
  a[4]=(_Float16)f1.x;  a[5]=(_Float16)f1.y;  a[6]=(_Float16)f1.z;  a[7]=(_Float16)f1.w;
  a[8]=(_Float16)f2.x;  a[9]=(_Float16)f2.y;  a[10]=(_Float16)f2.z; a[11]=(_Float16)f2.w;
  a[12]=(_Float16)f3.x; a[13]=(_Float16)f3.y; a[14]=(_Float16)f3.z; a[15]=(_Float16)f3.w;
  return a;
}

// A-fragment (16x32, MxK) from an f32 row: halfs at k0 = ks*32 + kg*8 and k0+16.
__device__ __forceinline__ v16h afrag_f32(const float* row, int k0) {
  return pack16(*(const float4*)(row + k0),      *(const float4*)(row + k0 + 4),
                *(const float4*)(row + k0 + 16), *(const float4*)(row + k0 + 20));
}

// B-fragment (32x16, KxN) for A@W^T from an f32 W row: 8 f32 at row and 8 at row+16.
__device__ __forceinline__ v16h bfrag_f32(const float* row) {
  return pack16(((const float4*)row)[0], ((const float4*)row)[1],
                ((const float4*)row)[4], ((const float4*)row)[5]);
}

// Concatenate two 8-half chunks (16B-aligned loads) into a v16h.
__device__ __forceinline__ v16h cat8(v8h lo, v8h hi) {
  v16h a;
#pragma unroll
  for (int e = 0; e < 8; ++e) { a[e] = lo[e]; a[8 + e] = hi[e]; }
  return a;
}

// A-fragment from an f16 row (two 8-half chunks at k0 and k0+16).
__device__ __forceinline__ v16h afrag_f16(const _Float16* row, int k0) {
  return cat8(*(const v8h*)(row + k0), *(const v8h*)(row + k0 + 16));
}

// B-fragment: two 8-half chunks at p and p+16, only 16B alignment guaranteed.
__device__ __forceinline__ v16h bfrag_f16(const _Float16* p) {
  return cat8(*(const v8h*)p, *(const v8h*)(p + 16));
}

#if USE_TDM
// Issue one TDM 2D tile load (global -> LDS), D# per CDNA5 ISA ch.8:
//   group0: count=1 | lds_addr | global_addr | type=2
//   group1: data_size=2B, pad_enable, pad_interval/amount, dims = tile dims
// d0 = row length (2B elements), d1 = rows, padint/padamt per ISA encoding.
__device__ __forceinline__ void tdm_load_2d(unsigned lds_off, const void* gptr,
                                            unsigned d0, unsigned d1,
                                            unsigned padint, unsigned padamt) {
  unsigned long long ga = (unsigned long long)(uintptr_t)gptr;
  v4u g0 = { 1u,                                   // count=1 (valid), no gather
             lds_off,                               // LDS byte offset
             (unsigned)ga,                          // global_addr[31:0]
             (unsigned)(ga >> 32) | (2u << 30) };   // addr[56:32] | type=2
  v8i g1;
  g1[0] = (int)((1u << 16) |                       // data_size = 2 bytes
                (1u << 20) |                       // pad_enable (LDS padding)
                (padint << 22) | (padamt << 25));
  g1[1] = (int)((d0 & 0xFFFFu) << 16);             // tensor_dim0[15:0]
  g1[2] = (int)((d0 >> 16) | ((d1 & 0xFFFFu) << 16));   // dim0[31:16]|dim1[15:0]
  g1[3] = (int)((d1 >> 16) | ((d0 & 0xFFFFu) << 16));   // dim1[31:16]|tile_dim0
  g1[4] = (int)(d1 & 0xFFFFu);                     // tile_dim1 (tile_dim2=0)
  g1[5] = (int)d0;                                 // tensor_dim0_stride[31:0]
  g1[6] = 0;                                       // stride0[47:32]|stride1 lo
  g1[7] = 0;
  v4i gz = {0, 0, 0, 0};
#if __clang_major__ >= 23
  v8i gz8 = {0, 0, 0, 0, 0, 0, 0, 0};
  __builtin_amdgcn_tensor_load_to_lds(g0, g1, gz, gz, gz8, 0);
#else
  __builtin_amdgcn_tensor_load_to_lds(g0, g1, gz, gz, 0);
#endif
}
#endif

// ---------------------------------------------------------------------------
// Kernel 1: q/k/v projections.  task -> (mtile, which-weight, 64-col group).
// q,k stored [B,H,T,D]; v stored transposed [B,H,D,T].
// ---------------------------------------------------------------------------
__global__ void qkv_proj_kernel(const float* __restrict__ x,
                                const float* __restrict__ Wq,
                                const float* __restrict__ Wk,
                                const float* __restrict__ Wv,
                                _Float16* __restrict__ q,
                                _Float16* __restrict__ k,
                                _Float16* __restrict__ vt) {
  const int lane  = threadIdx.x & 31;
  const int w     = threadIdx.x >> 5;
  const int kg    = lane >> 4;     // k-group (which half of the wave)
  const int nlane = lane & 15;

  int task  = blockIdx.x * 8 + w;        // 73728 wave tasks
  int mtile = task / 18;
  int rem   = task % 18;
  int wsel  = rem / 6;                   // 0=q, 1=k, 2=v
  int ng    = rem % 6;                   // 64-wide output column group == head
  int m0    = mtile * 16;
  int b     = m0 >> 8;
  int t0    = m0 & (T_ - 1);
  int n0    = ng * 64;

  const float* W = (wsel == 0) ? Wq : (wsel == 1 ? Wk : Wv);
  const float* xrow = x + (size_t)(m0 + nlane) * C_;

  v8f acc[4] = {{0,0,0,0,0,0,0,0}, {0,0,0,0,0,0,0,0},
                {0,0,0,0,0,0,0,0}, {0,0,0,0,0,0,0,0}};

#pragma unroll
  for (int ks = 0; ks < 12; ++ks) {
    int k0 = ks * 32 + kg * 8;
    if (ks < 11) __builtin_prefetch(xrow + k0 + 32, 0, 0);
    v16h a = afrag_f32(xrow, k0);
#pragma unroll
    for (int j = 0; j < 4; ++j) {
      const float* wrow = W + (size_t)(n0 + j * 16 + nlane) * C_ + ks * 32 + kg * 8;
      acc[j] = wmma_f16(a, bfrag_f32(wrow), acc[j]);
    }
  }

  if (wsel < 2) {
    _Float16* dst = (wsel == 0) ? q : k;
#pragma unroll
    for (int j = 0; j < 4; ++j) {
      int n = n0 + j * 16 + nlane;
      int h = n >> 6, d = n & 63;
#pragma unroll
      for (int r = 0; r < 8; ++r) {
        int t = t0 + 8 * kg + r;
        dst[(size_t)(((b * H_ + h) * T_) + t) * D_ + d] = (_Float16)acc[j][r];
      }
    }
  } else {
    // vT: [B,H,D,T]; this n-group is exactly head ng; 8 consecutive t per lane.
#pragma unroll
    for (int j = 0; j < 4; ++j) {
      int d = j * 16 + nlane;
      v8h pk;
#pragma unroll
      for (int r = 0; r < 8; ++r) pk[r] = (_Float16)acc[j][r];
      *(v8h*)(vt + (size_t)(((b * H_ + ng) * D_) + d) * T_ + t0 + 8 * kg) = pk;
    }
  }
}

// ---------------------------------------------------------------------------
// Kernel 2: causal attention, one block per (b,h); 4 waves, balanced row-tiles.
// K and V^T tiles staged into LDS via the Tensor Data Mover.
// ---------------------------------------------------------------------------
__global__ void attn_kernel(const _Float16* __restrict__ q,
                            const _Float16* __restrict__ k,
                            const _Float16* __restrict__ vt,
                            _Float16* __restrict__ ao) {
  __shared__ _Float16 Ks[T_ * RSTK];      // 36864 B (padded rows)
  __shared__ _Float16 Vts[D_ * RSTV];     // 33792 B (padded rows)
  __shared__ _Float16 Ps[4][16 * PSTR];   // 33792 B per-wave P transpose

  const int lane  = threadIdx.x & 31;
  const int w     = threadIdx.x >> 5;
  const int kg    = lane >> 4;
  const int nlane = lane & 15;

  const int bh = blockIdx.x;
  const int b  = bh / H_;
  const int h  = bh % H_;
  const _Float16* Qp  = q  + (size_t)bh * T_ * D_;
  const _Float16* Kp  = k  + (size_t)bh * T_ * D_;
  const _Float16* Vtp = vt + (size_t)bh * D_ * T_;
  _Float16* Pw = &Ps[w][0];
  const float NEG_INF = -__builtin_huge_valf();

  // ---- stage K [256x64] and V^T [64x256] tiles into padded LDS ----
#if USE_TDM
  if (w == 0) {
    // K: 32-dword rows, pad 4 dwords  -> interval code 4 (32), amount code 3 (4)
    tdm_load_2d((unsigned)(uintptr_t)&Ks[0],  Kp,  D_, T_, 4u, 3u);
    // V^T: 128-dword rows, pad 4 dwords -> interval code 6 (128), amount code 3
    tdm_load_2d((unsigned)(uintptr_t)&Vts[0], Vtp, T_, D_, 6u, 3u);
    __builtin_amdgcn_s_wait_tensorcnt(0);
  }
  __syncthreads();
#else
  // Fallback: cooperative copy replicating the padded layout.
  for (int i = threadIdx.x; i < T_ * (D_ / 8); i += 128) {   // 2048 x 16B
    int row = i >> 3, c8 = i & 7;
    *(uint4*)(Ks + row * RSTK + c8 * 8) =
        *(const uint4*)(Kp + (size_t)row * D_ + c8 * 8);
  }
  for (int i = threadIdx.x; i < D_ * (T_ / 8); i += 128) {   // 2048 x 16B
    int row = i >> 5, c8 = i & 31;
    *(uint4*)(Vts + row * RSTV + c8 * 8) =
        *(const uint4*)(Vtp + (size_t)row * T_ + c8 * 8);
  }
  __syncthreads();
#endif

  const int rts[4] = {w, 7 - w, 8 + w, 15 - w};   // constant work per wave

#pragma unroll 1
  for (int it = 0; it < 4; ++it) {
    const int rt = rts[it];
    const int m0 = rt * 16;

    // ---- S = (Q K^T) row-tile: tiles j = 0..rt (rest fully causal-masked) ----
    v8f s[16];
#pragma unroll
    for (int j = 0; j < 16; ++j) s[j] = (v8f){0,0,0,0,0,0,0,0};

    v16h aq0 = afrag_f16(Qp + (size_t)(m0 + nlane) * D_, 0  + kg * 8);
    v16h aq1 = afrag_f16(Qp + (size_t)(m0 + nlane) * D_, 32 + kg * 8);
#pragma unroll
    for (int j = 0; j < 16; ++j)
      if (j <= rt) {
        const _Float16* kr = Ks + (j * 16 + nlane) * RSTK;
        s[j] = wmma_f16(aq0, bfrag_f16(kr + kg * 8), s[j]);
        s[j] = wmma_f16(aq1, bfrag_f16(kr + 32 + kg * 8), s[j]);
      }

    // ---- softmax over rows (rows live in one 16-lane half, vgpr index r) ----
#pragma unroll
    for (int r = 0; r < 8; ++r) {
      const int m = m0 + 8 * kg + r;
      float mx = NEG_INF;
#pragma unroll
      for (int j = 0; j < 16; ++j)
        if (j <= rt) {
          float v = s[j][r] * 0.125f;                 // 1/sqrt(64)
          if (j * 16 + nlane > m) v = NEG_INF;        // causal (diag tile only)
          s[j][r] = v;
          mx = fmaxf(mx, v);
        }
      mx = fmaxf(mx, __shfl_xor(mx, 1));
      mx = fmaxf(mx, __shfl_xor(mx, 2));
      mx = fmaxf(mx, __shfl_xor(mx, 4));
      mx = fmaxf(mx, __shfl_xor(mx, 8));
      float sum = 0.f;
#pragma unroll
      for (int j = 0; j < 16; ++j)
        if (j <= rt) {
          float e = __expf(s[j][r] - mx);
          s[j][r] = e;
          sum += e;
        }
      sum += __shfl_xor(sum, 1);
      sum += __shfl_xor(sum, 2);
      sum += __shfl_xor(sum, 4);
      sum += __shfl_xor(sum, 8);
      float inv = 1.0f / sum;
      // transpose P through LDS (row-major 16 x 256, padded stride)
#pragma unroll
      for (int j = 0; j < 16; ++j)
        if (j <= rt)
          Pw[(8 * kg + r) * PSTR + j * 16 + nlane] = (_Float16)(s[j][r] * inv);
      if ((rt & 1) == 0)   // zero the half K-step that the PV loop still touches
        Pw[(8 * kg + r) * PSTR + (rt + 1) * 16 + nlane] = (_Float16)0.f;
    }

    // ---- O = P V  (K runs only over the unmasked prefix) ----
    v8f o[4] = {{0,0,0,0,0,0,0,0}, {0,0,0,0,0,0,0,0},
                {0,0,0,0,0,0,0,0}, {0,0,0,0,0,0,0,0}};
    const int ktmax = rt >> 1;
#pragma unroll
    for (int kt = 0; kt < 8; ++kt)
      if (kt <= ktmax) {
        v16h ap = afrag_f16(Pw + nlane * PSTR, kt * 32 + kg * 8);
#pragma unroll
        for (int j = 0; j < 4; ++j) {
          const _Float16* vr = Vts + (j * 16 + nlane) * RSTV;
          o[j] = wmma_f16(ap, bfrag_f16(vr + kt * 32 + kg * 8), o[j]);
        }
      }

    // ---- store merged-head layout [B*T, C] f16 ----
#pragma unroll
    for (int j = 0; j < 4; ++j) {
      int d = j * 16 + nlane;
#pragma unroll
      for (int r = 0; r < 8; ++r) {
        int t = m0 + 8 * kg + r;
        ao[(size_t)(b * T_ + t) * C_ + h * D_ + d] = (_Float16)o[j][r];
      }
    }
  }
}

// ---------------------------------------------------------------------------
// Kernel 3: out = ao @ Wp^T + bp  (f32 output)
// ---------------------------------------------------------------------------
__global__ void out_proj_kernel(const _Float16* __restrict__ ao,
                                const float* __restrict__ Wp,
                                const float* __restrict__ bp,
                                float* __restrict__ out) {
  const int lane  = threadIdx.x & 31;
  const int w     = threadIdx.x >> 5;
  const int kg    = lane >> 4;
  const int nlane = lane & 15;

  int task  = blockIdx.x * 8 + w;     // 24576 wave tasks
  int mtile = task / 6;
  int ng    = task % 6;
  int m0    = mtile * 16;
  int n0    = ng * 64;

  const _Float16* arow = ao + (size_t)(m0 + nlane) * C_;

  v8f acc[4] = {{0,0,0,0,0,0,0,0}, {0,0,0,0,0,0,0,0},
                {0,0,0,0,0,0,0,0}, {0,0,0,0,0,0,0,0}};

#pragma unroll
  for (int ks = 0; ks < 12; ++ks) {
    int k0 = ks * 32 + kg * 8;
    if (ks < 11) __builtin_prefetch(arow + k0 + 32, 0, 0);
    v16h a = afrag_f16(arow, k0);
#pragma unroll
    for (int j = 0; j < 4; ++j) {
      const float* wrow = Wp + (size_t)(n0 + j * 16 + nlane) * C_ + ks * 32 + kg * 8;
      acc[j] = wmma_f16(a, bfrag_f32(wrow), acc[j]);
    }
  }

#pragma unroll
  for (int j = 0; j < 4; ++j) {
    int n = n0 + j * 16 + nlane;
    float bias = bp[n];
#pragma unroll
    for (int r = 0; r < 8; ++r)
      out[(size_t)(m0 + 8 * kg + r) * C_ + n] = acc[j][r] + bias;
  }
}

// ---------------------------------------------------------------------------
extern "C" void kernel_launch(void* const* d_in, const int* in_sizes, int n_in,
                              void* d_out, int out_size, void* d_ws, size_t ws_size,
                              hipStream_t stream) {
  // setup_inputs() order: x, Wk, Wq, Wv, Wp, bp
  const float* x  = (const float*)d_in[0];
  const float* Wk = (const float*)d_in[1];
  const float* Wq = (const float*)d_in[2];
  const float* Wv = (const float*)d_in[3];
  const float* Wp = (const float*)d_in[4];
  const float* bp = (const float*)d_in[5];
  float* out = (float*)d_out;

  const size_t BUF = (size_t)BT_ * C_ * sizeof(_Float16);   // 48 MB each
  char* ws = (char*)d_ws;
  _Float16* q  = (_Float16*)(ws);
  _Float16* k  = (_Float16*)(ws + BUF);
  _Float16* vt = (_Float16*)(ws + 2 * BUF);
  _Float16* ao = (_Float16*)(ws + 3 * BUF);   // needs ws_size >= 192 MB

  qkv_proj_kernel<<<9216, 256, 0, stream>>>(x, Wq, Wk, Wv, q, k, vt);
  attn_kernel<<<B_ * H_, 128, 0, stream>>>(q, k, vt, ao);
  out_proj_kernel<<<3072, 256, 0, stream>>>(ao, Wp, bp, out);
}
